// PointerNet_69638599737573
// MI455X (gfx1250) — matrix operand order repaired and checked by reading, not compiled, not run on hardware
//
#include <hip/hip_runtime.h>

#define B_DIM 64
#define L_DIM 2048
#define H_DIM 256
#define ROWS  128      // L-rows per block (8 waves x 16 rows)
#define WPAD  264      // padded LDS row length (bf16 elems) -> 528B rows, conflict-free

typedef __attribute__((ext_vector_type(16))) __bf16       v16bf;
typedef __attribute__((ext_vector_type(8)))  float        v8f;
typedef __attribute__((ext_vector_type(4)))  unsigned int u32x4;
typedef __attribute__((ext_vector_type(4)))  int          i32x4;

union FragU { v16bf v; u32x4 u[2]; };

#if __has_builtin(__builtin_amdgcn_global_load_async_to_lds_b128) && \
    __has_builtin(__builtin_amdgcn_s_wait_asynccnt)
#define USE_ASYNC_LDS 1
typedef __attribute__((address_space(1))) i32x4 g_i32x4;   // global AS
typedef __attribute__((address_space(3))) i32x4 l_i32x4;   // LDS AS
#endif

__device__ inline unsigned short f2bf(float f) {
    unsigned int u = __builtin_bit_cast(unsigned int, f);
    u += 0x7FFFu + ((u >> 16) & 1u);        // round-to-nearest-even
    return (unsigned short)(u >> 16);
}

// Branch-free tanh: 1 - 2/(e^{2x}+1). Saturates to +/-1 for large |x| in fp32;
// avoids libm tanhf's s_and_saveexec divergent region inside the WMMA loop.
__device__ inline float fast_tanh(float x) {
    const float e = __expf(2.0f * x);
    return 1.0f - __fdividef(2.0f, e + 1.0f);
}

// ---------------------------------------------------------------------------
// Convert w_u[:, H:] (element [d][H+h] at d*2H + H + h) to bf16 [d][h].
__global__ __launch_bounds__(256)
void wconv_kernel(const float* __restrict__ w_u, unsigned short* __restrict__ Wt) {
    const int d = blockIdx.x;
    const int h = threadIdx.x;
    Wt[d * H_DIM + h] = f2bf(w_u[d * (2 * H_DIM) + H_DIM + h]);
}

// ---------------------------------------------------------------------------
// sstate[b][d] = sum_h X[b][h] * w_u[d][h]   (first-half columns of w_u)
__global__ __launch_bounds__(256)
void state_proj_kernel(const float* __restrict__ X, const float* __restrict__ w_u,
                       float* __restrict__ sstate) {
    __shared__ float xs[H_DIM];
    const int b = blockIdx.x;
    const int d = threadIdx.x;
    xs[d] = X[b * H_DIM + d];
    __syncthreads();
    float acc = 0.f;
    #pragma unroll 8
    for (int h = 0; h < H_DIM; ++h) acc += xs[h] * w_u[d * (2 * H_DIM) + h];
    sstate[b * H_DIM + d] = acc;
}

// ---------------------------------------------------------------------------
// Fused pointer-score GEMM:
//   sraw[b][l] = sum_d w_s[d]*tanh(sstate[b][d] + sum_h match[b][l][h]*Wt[d][h])
//              + 1e30*(mask[b][l]-1)
__global__ __launch_bounds__(256)
void pointer_scores_kernel(const float* __restrict__ match,
                           const unsigned short* __restrict__ Wt,   // bf16 [256][256]
                           const float* __restrict__ sstate,        // [B][256]
                           const float* __restrict__ w_s,           // [256]
                           const float* __restrict__ mask,          // [B][L]
                           float* __restrict__ sraw) {               // [B][L]
    extern __shared__ char smem[];
    unsigned short* ldsW = (unsigned short*)smem;                       // [256][WPAD]
    unsigned short* ldsM = (unsigned short*)(smem + H_DIM * WPAD * 2);  // [ROWS][WPAD]

    const int b   = blockIdx.y;
    const int l0  = blockIdx.x * ROWS;
    const int tid = threadIdx.x;

    // Stage full bf16 weight matrix into LDS.
#ifdef USE_ASYNC_LDS
    // Async DMA path: 16B chunks, ASYNCcnt-tracked, no VGPR round-trip.
    for (int i = tid; i < H_DIM * (H_DIM / 8); i += 256) {
        const int row = i >> 5;          // 32 chunks of 8 ushorts per row
        const int c8  = (i & 31) << 3;
        __builtin_amdgcn_global_load_async_to_lds_b128(
            (g_i32x4*)(Wt + row * H_DIM + c8),
            (l_i32x4*)(ldsW + row * WPAD + c8),
            0, 0);
    }
#else
    for (int i = tid; i < H_DIM * (H_DIM / 4); i += 256) {
        const int row = i >> 6;
        const int c4  = (i & 63) << 2;
        *reinterpret_cast<uint2*>(ldsW + row * WPAD + c4) =
            *reinterpret_cast<const uint2*>(Wt + row * H_DIM + c4);
    }
#endif
    // Stage match tile: fp32 -> bf16 into LDS (conversion must go through VGPRs).
    const float* mb = match + ((size_t)b * L_DIM + l0) * H_DIM;
    for (int i = tid; i < ROWS * (H_DIM / 4); i += 256) {
        const int row = i >> 6;
        const int c4  = (i & 63) << 2;
        float4 f = *reinterpret_cast<const float4*>(mb + row * H_DIM + c4);
        uint2 p;
        p.x = (unsigned)f2bf(f.x) | ((unsigned)f2bf(f.y) << 16);
        p.y = (unsigned)f2bf(f.z) | ((unsigned)f2bf(f.w) << 16);
        *reinterpret_cast<uint2*>(ldsM + row * WPAD + c4) = p;
    }
#ifdef USE_ASYNC_LDS
    __builtin_amdgcn_s_wait_asynccnt(0);  // own async loads landed; barrier covers peers
#endif
    __syncthreads();

    const int wave    = tid >> 5;
    const int lane    = tid & 31;
    const int lhalf   = lane >> 4;   // 0: lanes 0-15, 1: lanes 16-31
    const int l16     = lane & 15;
    const int rowBase = wave * 16;

    // Preload 8 A fragments (16x32 bf16): lane=M, lanes<16 hold K {0..7,16..23}.
    FragU afrag[8];
    const unsigned short* arow = ldsM + (rowBase + l16) * WPAD + lhalf * 8;
    #pragma unroll
    for (int kt = 0; kt < 8; ++kt) {
        const unsigned short* p = arow + kt * 32;
        afrag[kt].u[0] = *reinterpret_cast<const u32x4*>(p);
        afrag[kt].u[1] = *reinterpret_cast<const u32x4*>(p + 16);
    }

    // Hoist per-lane w_s / sstate values for all 16 d-tiles out of the hot loop.
    float wsv[16], ssv[16];
    #pragma unroll
    for (int dt = 0; dt < 16; ++dt) {
        wsv[dt] = w_s[dt * 16 + l16];
        ssv[dt] = sstate[b * H_DIM + dt * 16 + l16];
    }

    float partial[8] = {0.f, 0.f, 0.f, 0.f, 0.f, 0.f, 0.f, 0.f};
    // Fully unrolled: no register-indexed (movrels) array access, and the
    // scheduler can overlap tile dt's tanh TRANS chain with tile dt+1's
    // ds_load + v_wmma stream.
    #pragma unroll
    for (int dt = 0; dt < 16; ++dt) {
        const int dcol = dt * 16 + l16;
        // B fragment (32x16): lane=N, lanes<16 hold K {0..7,16..23}, lanes>=16 hold K {8..15,24..31}.
        const unsigned short* brow = ldsW + dcol * WPAD + lhalf * 8;
        v8f acc = {0.f, 0.f, 0.f, 0.f, 0.f, 0.f, 0.f, 0.f};
        #pragma unroll
        for (int kt = 0; kt < 8; ++kt) {
            FragU bfrag;
            const unsigned short* q = brow + kt * 32;
            bfrag.u[0] = *reinterpret_cast<const u32x4*>(q);
            bfrag.u[1] = *reinterpret_cast<const u32x4*>(q + 16);
            acc = __builtin_amdgcn_wmma_f32_16x16x32_bf16(
                      false, afrag[kt].v, false, bfrag.v, (short)0, acc, false, false);
        }
        // C layout: lane=N(col d), VGPR i = row (lanes>=16: rows 8..15).
        #pragma unroll
        for (int i = 0; i < 8; ++i) partial[i] += wsv[dt] * fast_tanh(ssv[dt] + acc[i]);
    }
    // Reduce over the 16 d-columns held by each lane-group.
    #pragma unroll
    for (int off = 1; off < 16; off <<= 1) {
        #pragma unroll
        for (int i = 0; i < 8; ++i) partial[i] += __shfl_xor(partial[i], off, 32);
    }
    if (l16 == 0) {
        const int rb = rowBase + lhalf * 8;
        #pragma unroll
        for (int i = 0; i < 8; ++i) {
            const int   l  = l0 + rb + i;
            const float mv = mask[b * L_DIM + l];
            sraw[(size_t)b * L_DIM + l] = partial[i] + 1.0e30f * (mv - 1.0f);
        }
    }
}

// ---------------------------------------------------------------------------
// Per-batch: softmax(s1) -> logits1, attention-weighted sum of match -> res,
// then fused GRU step -> state.  Block = one batch row, 256 threads.
__global__ __launch_bounds__(256)
void softmax_res_gru_kernel(const float* __restrict__ sraw, const float* __restrict__ match,
                            const float* __restrict__ init, const float* __restrict__ w_ih,
                            const float* __restrict__ w_hh, const float* __restrict__ b_ih,
                            const float* __restrict__ b_hh, float* __restrict__ logits1,
                            float* __restrict__ state) {
    __shared__ float red[256];
    __shared__ float aS[L_DIM];
    __shared__ float resS[H_DIM];
    __shared__ float initS[H_DIM];
    const int b = blockIdx.x;
    const int t = threadIdx.x;

    float m = -3.0e38f;
    for (int i = t; i < L_DIM; i += 256) {
        const float v = sraw[(size_t)b * L_DIM + i];
        aS[i] = v;
        m = fmaxf(m, v);
    }
    red[t] = m;
    __syncthreads();
    for (int s = 128; s > 0; s >>= 1) { if (t < s) red[t] = fmaxf(red[t], red[t + s]); __syncthreads(); }
    m = red[0];
    __syncthreads();

    float sum = 0.f;
    for (int i = t; i < L_DIM; i += 256) {
        const float e = __expf(aS[i] - m);
        aS[i] = e;
        sum += e;
    }
    red[t] = sum;
    __syncthreads();
    for (int s = 128; s > 0; s >>= 1) { if (t < s) red[t] += red[t + s]; __syncthreads(); }
    const float inv = 1.0f / red[0];
    __syncthreads();
    for (int i = t; i < L_DIM; i += 256) {
        const float a = aS[i] * inv;
        aS[i] = a;
        logits1[(size_t)b * L_DIM + i] = a;
    }
    __syncthreads();

    // res[h] = sum_l a[l] * match[b][l][h]   (thread t = h, coalesced match reads)
    float acc = 0.f;
    const float* mbp = match + (size_t)b * L_DIM * H_DIM;
    #pragma unroll 8
    for (int l = 0; l < L_DIM; ++l) acc += aS[l] * mbp[l * H_DIM + t];
    resS[t]  = acc;
    initS[t] = init[b * H_DIM + t];
    __syncthreads();

    // GRU gates for hidden index t
    float gi_r = b_ih[t],           gh_r = b_hh[t];
    float gi_z = b_ih[H_DIM + t],   gh_z = b_hh[H_DIM + t];
    float gi_n = b_ih[2*H_DIM + t], gh_n = b_hh[2*H_DIM + t];
    #pragma unroll 4
    for (int h = 0; h < H_DIM; ++h) {
        const float x  = resS[h];
        const float hp = initS[h];
        gi_r += x  * w_ih[(t)            * H_DIM + h];
        gi_z += x  * w_ih[(H_DIM + t)    * H_DIM + h];
        gi_n += x  * w_ih[(2*H_DIM + t)  * H_DIM + h];
        gh_r += hp * w_hh[(t)            * H_DIM + h];
        gh_z += hp * w_hh[(H_DIM + t)    * H_DIM + h];
        gh_n += hp * w_hh[(2*H_DIM + t)  * H_DIM + h];
    }
    const float r = 1.f / (1.f + __expf(-(gi_r + gh_r)));
    const float z = 1.f / (1.f + __expf(-(gi_z + gh_z)));
    const float n = fast_tanh(gi_n + r * gh_n);
    state[b * H_DIM + t] = (1.f - z) * n + z * initS[t];
}

// ---------------------------------------------------------------------------
__global__ __launch_bounds__(256)
void softmax_kernel(const float* __restrict__ sraw, float* __restrict__ out) {
    __shared__ float red[256];
    __shared__ float aS[L_DIM];
    const int b = blockIdx.x;
    const int t = threadIdx.x;
    float m = -3.0e38f;
    for (int i = t; i < L_DIM; i += 256) {
        const float v = sraw[(size_t)b * L_DIM + i];
        aS[i] = v;
        m = fmaxf(m, v);
    }
    red[t] = m;
    __syncthreads();
    for (int s = 128; s > 0; s >>= 1) { if (t < s) red[t] = fmaxf(red[t], red[t + s]); __syncthreads(); }
    m = red[0];
    __syncthreads();
    float sum = 0.f;
    for (int i = t; i < L_DIM; i += 256) {
        const float e = __expf(aS[i] - m);
        aS[i] = e;
        sum += e;
    }
    red[t] = sum;
    __syncthreads();
    for (int s = 128; s > 0; s >>= 1) { if (t < s) red[t] += red[t + s]; __syncthreads(); }
    const float inv = 1.0f / red[0];
    __syncthreads();
    for (int i = t; i < L_DIM; i += 256) out[(size_t)b * L_DIM + i] = aS[i] * inv;
}

// ---------------------------------------------------------------------------
extern "C" void kernel_launch(void* const* d_in, const int* in_sizes, int n_in,
                              void* d_out, int out_size, void* d_ws, size_t ws_size,
                              hipStream_t stream) {
    const float* init  = (const float*)d_in[0];
    const float* match = (const float*)d_in[1];
    const float* mask  = (const float*)d_in[2];
    const float* w_u1  = (const float*)d_in[3];
    const float* w_s1  = (const float*)d_in[4];
    const float* w_u2  = (const float*)d_in[5];
    const float* w_s2  = (const float*)d_in[6];
    const float* w_ih  = (const float*)d_in[7];
    const float* w_hh  = (const float*)d_in[8];
    const float* b_ih  = (const float*)d_in[9];
    const float* b_hh  = (const float*)d_in[10];
    float* out = (float*)d_out;   // logits1 [0, B*L), logits2 [B*L, 2*B*L)

    char* w = (char*)d_ws;
    unsigned short* Wt1 = (unsigned short*)w; w += (size_t)H_DIM * H_DIM * 2;
    unsigned short* Wt2 = (unsigned short*)w; w += (size_t)H_DIM * H_DIM * 2;
    float* ss1   = (float*)w; w += (size_t)B_DIM * H_DIM * 4;
    float* ss2   = (float*)w; w += (size_t)B_DIM * H_DIM * 4;
    float* s1raw = (float*)w; w += (size_t)B_DIM * L_DIM * 4;
    float* s2raw = (float*)w; w += (size_t)B_DIM * L_DIM * 4;
    float* state = (float*)w; w += (size_t)B_DIM * H_DIM * 4;

    const size_t shA = (size_t)(H_DIM + ROWS) * WPAD * 2;  // 202752 B dynamic LDS
    const dim3 gridA(L_DIM / ROWS, B_DIM);

    wconv_kernel<<<H_DIM, 256, 0, stream>>>(w_u1, Wt1);
    wconv_kernel<<<H_DIM, 256, 0, stream>>>(w_u2, Wt2);
    state_proj_kernel<<<B_DIM, 256, 0, stream>>>(init, w_u1, ss1);
    pointer_scores_kernel<<<gridA, 256, shA, stream>>>(match, Wt1, ss1, w_s1, mask, s1raw);
    softmax_res_gru_kernel<<<B_DIM, 256, 0, stream>>>(s1raw, match, init, w_ih, w_hh,
                                                      b_ih, b_hh, out, state);
    state_proj_kernel<<<B_DIM, 256, 0, stream>>>(state, w_u2, ss2);
    pointer_scores_kernel<<<gridA, 256, shA, stream>>>(match, Wt2, ss2, w_s2, mask, s2raw);
    softmax_kernel<<<B_DIM, 256, 0, stream>>>(s2raw, out + (size_t)B_DIM * L_DIM);
}
